// Net2_86397562127201
// MI455X (gfx1250) — matrix operand order repaired and checked by reading, not compiled, not run on hardware
//
#include <hip/hip_runtime.h>

// ---------------------------------------------------------------------------
// GCNConv(x @ W1, deg-normalized scatter-add) + ReLU MLP head, CDNA5 (gfx1250)
//
// Design for MI455X:
//  * All 3 GEMMs via v_wmma_f32_16x16x32_bf16 (bf16 in, f32 accum).
//  * h is stored bf16 (60.8 MB) so h + f32 agg (121.6 MB) = 182.4 MB fits the
//    192 MB L2 -> the 800k-edge gather + 240M f32 atomics run entirely in L2.
//  * K dims padded to multiples of 32 (zeros), N dims to multiples of 16.
// ---------------------------------------------------------------------------

typedef __attribute__((ext_vector_type(16))) __bf16 v16bf;
typedef __attribute__((ext_vector_type(8)))  float  v8f;

#define N_NODES 100000
#define N_EDGES 800000
#define F_IN    128
#define D1      300
#define D1P     304      // h/agg feature stride (19 tiles of 16)
#define D1K     320      // K-padded stride for aggb (10 K-blocks of 32)
#define D2      100
#define D2P     128      // padded D2 (8 tiles / 4 K-blocks)
#define MTILES  (N_NODES / 16)   // 6250

// ---- WMMA fragment loaders (layouts per CDNA5 ISA 7.12.2, wave32) ----------

// A: 16x32 bf16. lane L holds row M = L&15; lanes 0-15: K = k0+0..7,16..23;
// lanes 16-31: K = k0+8..15,24..31. Two 16B loads per fragment.
static __device__ __forceinline__ v16bf load_a_frag(const __bf16* A, int lda,
                                                    int m, int k0, int lane) {
  int kb = k0 + ((lane >> 4) << 3);
  const uint4* p = (const uint4*)(A + (size_t)m * lda + kb);
  union { uint4 q[2]; v16bf v; } u;
  u.q[0] = p[0];   // K = kb .. kb+7
  u.q[1] = p[2];   // K = kb+16 .. kb+23
  return u.v;
}

// B: 32x16 bf16. lane L holds column N = L&15; lanes 0-15: K = k0+0..7,16..23;
// lanes 16-31: K = k0+8..15,24..31 (same K order as A).
static __device__ __forceinline__ v16bf load_b_frag(const __bf16* B, int ldb,
                                                    int k0, int n0, int lane) {
  int n = n0 + (lane & 15);
  int kb = k0 + ((lane >> 4) << 3);
  v16bf v;
#pragma unroll
  for (int s = 0; s < 8; ++s) {
    v[s]     = B[(size_t)(kb + s) * ldb + n];
    v[s + 8] = B[(size_t)(kb + 16 + s) * ldb + n];
  }
  return v;
}

// ---- setup kernels ---------------------------------------------------------

__global__ void k_init_deg(float* deg) {
  int i = blockIdx.x * blockDim.x + threadIdx.x;
  if (i < N_NODES) deg[i] = 1.0f;              // self-loop
}

__global__ void k_deg_accum(const long long* __restrict__ ei, float* deg) {
  int e = blockIdx.x * blockDim.x + threadIdx.x;
  if (e < N_EDGES) atomicAdd(&deg[(int)ei[N_EDGES + e]], 1.0f);  // dst row
}

__global__ void k_dinv(const float* __restrict__ deg, float* dinv) {
  int i = blockIdx.x * blockDim.x + threadIdx.x;
  if (i < N_NODES) dinv[i] = rsqrtf(deg[i]);
}

__global__ void k_cvt_x(const float* __restrict__ x, __bf16* __restrict__ xb) {
  int i = blockIdx.x * blockDim.x + threadIdx.x;
  if (i < N_NODES * F_IN) xb[i] = (__bf16)x[i];
}

__global__ void k_cvt_W1(const float* __restrict__ W, __bf16* __restrict__ Wb) {
  int i = blockIdx.x * blockDim.x + threadIdx.x;        // 128 x 320
  if (i >= F_IN * D1K) return;
  int k = i / D1K, n = i % D1K;
  Wb[i] = (n < D1) ? (__bf16)W[k * D1 + n] : (__bf16)0.0f;
}

__global__ void k_cvt_Wf1(const float* __restrict__ W, __bf16* __restrict__ Wb) {
  int i = blockIdx.x * blockDim.x + threadIdx.x;        // 320 x 128
  if (i >= D1K * D2P) return;
  int k = i / D2P, n = i % D2P;
  Wb[i] = (k < D1 && n < D2) ? (__bf16)W[k * D2 + n] : (__bf16)0.0f;
}

__global__ void k_cvt_Wf2(const float* __restrict__ W, __bf16* __restrict__ Wb) {
  int i = blockIdx.x * blockDim.x + threadIdx.x;        // 128 x 128
  if (i >= D2P * F_IN) return;
  int k = i / F_IN, n = i % F_IN;
  Wb[i] = (k < D2) ? (__bf16)W[k * F_IN + n] : (__bf16)0.0f;
}

__global__ void k_zero_agg(float* agg) {
  size_t i = (size_t)blockIdx.x * blockDim.x + threadIdx.x;
  if (i < (size_t)N_NODES * D1P) agg[i] = 0.0f;
}

// ---- GEMM 1: h = x @ W1  (bf16 out, stride D1P) ----------------------------

__global__ void k_gemm1(const __bf16* __restrict__ xb,
                        const __bf16* __restrict__ W1b,
                        __bf16* __restrict__ h) {
  int lane = threadIdx.x;
  int mt = blockIdx.x * blockDim.y + threadIdx.y;   // wave-uniform
  if (mt >= MTILES) return;
  int m0 = mt * 16;
  int n0 = blockIdx.y * 16;                          // 19 tiles -> cols 0..303
  v8f c = {};
#pragma unroll
  for (int k0 = 0; k0 < F_IN; k0 += 32) {
    v16bf a = load_a_frag(xb, F_IN, m0 + (lane & 15), k0, lane);
    v16bf b = load_b_frag(W1b, D1K, k0, n0, lane);
    c = __builtin_amdgcn_wmma_f32_16x16x32_bf16(false, a, false, b,
                                                (short)0, c, false, false);
  }
  int n  = n0 + (lane & 15);
  int mB = m0 + ((lane >> 4) << 3);
#pragma unroll
  for (int r = 0; r < 8; ++r)
    h[(size_t)(mB + r) * D1P + n] = (__bf16)c[r];
}

// ---- edge scatter: agg[dst] += h[src] * dinv[src]*dinv[dst]  (L2-resident) -

__global__ void k_scatter(const long long* __restrict__ ei,
                          const __bf16* __restrict__ h,
                          const float* __restrict__ dinv,
                          float* __restrict__ agg) {
  int e = blockIdx.x * (blockDim.x >> 5) + (threadIdx.x >> 5);
  if (e >= N_EDGES) return;
  int lane = threadIdx.x & 31;
  int s = (int)ei[e];
  int d = (int)ei[N_EDGES + e];
  float w = dinv[s] * dinv[d];
  const __bf16* hs = h + (size_t)s * D1P;
  float* ad = agg + (size_t)d * D1P;
  for (int f = lane; f < D1; f += 32)
    atomicAdd(&ad[f], (float)hs[f] * w);
}

// ---- epilogue 1: self-loop + bias + relu, emit K-padded bf16 activations ---

__global__ void k_epi1(const float* __restrict__ agg,
                       const __bf16* __restrict__ h,
                       const float* __restrict__ dinv,
                       const float* __restrict__ b1,
                       __bf16* __restrict__ aggb) {
  int i = blockIdx.x;
  int f = threadIdx.x;                               // blockDim.x == 320
  float v = 0.0f;
  if (f < D1) {
    float di = dinv[i];
    v = agg[(size_t)i * D1P + f] + (float)h[(size_t)i * D1P + f] * di * di + b1[f];
    v = fmaxf(v, 0.0f);
  }
  aggb[(size_t)i * D1K + f] = (__bf16)v;             // cols 300..319 -> 0
}

// ---- GEMM 2: h2 = relu(aggb @ Wf1 + bf1)  (bf16 out, stride D2P) -----------

__global__ void k_gemm2(const __bf16* __restrict__ aggb,
                        const __bf16* __restrict__ Wf1b,
                        const float* __restrict__ bf1,
                        __bf16* __restrict__ h2b) {
  int lane = threadIdx.x;
  int mt = blockIdx.x * blockDim.y + threadIdx.y;
  if (mt >= MTILES) return;
  int m0 = mt * 16;
  int n0 = blockIdx.y * 16;                          // 8 tiles
  v8f c = {};
#pragma unroll
  for (int k0 = 0; k0 < D1K; k0 += 32) {
    v16bf a = load_a_frag(aggb, D1K, m0 + (lane & 15), k0, lane);
    v16bf b = load_b_frag(Wf1b, D2P, k0, n0, lane);
    c = __builtin_amdgcn_wmma_f32_16x16x32_bf16(false, a, false, b,
                                                (short)0, c, false, false);
  }
  int n  = n0 + (lane & 15);
  int mB = m0 + ((lane >> 4) << 3);
  float bias = (n < D2) ? bf1[n] : 0.0f;
#pragma unroll
  for (int r = 0; r < 8; ++r) {
    float v = fmaxf(c[r] + bias, 0.0f);
    h2b[(size_t)(mB + r) * D2P + n] = (__bf16)v;     // cols >= 100 stay 0
  }
}

// ---- GEMM 3: out = h2 @ Wf2 + bf2  (f32 out, stride F_IN) ------------------

__global__ void k_gemm3(const __bf16* __restrict__ h2b,
                        const __bf16* __restrict__ Wf2b,
                        const float* __restrict__ bf2,
                        float* __restrict__ out) {
  int lane = threadIdx.x;
  int mt = blockIdx.x * blockDim.y + threadIdx.y;
  if (mt >= MTILES) return;
  int m0 = mt * 16;
  int n0 = blockIdx.y * 16;                          // 8 tiles
  v8f c = {};
#pragma unroll
  for (int k0 = 0; k0 < D2P; k0 += 32) {
    v16bf a = load_a_frag(h2b, D2P, m0 + (lane & 15), k0, lane);
    v16bf b = load_b_frag(Wf2b, F_IN, k0, n0, lane);
    c = __builtin_amdgcn_wmma_f32_16x16x32_bf16(false, a, false, b,
                                                (short)0, c, false, false);
  }
  int n  = n0 + (lane & 15);
  int mB = m0 + ((lane >> 4) << 3);
  float bias = bf2[n];
#pragma unroll
  for (int r = 0; r < 8; ++r)
    out[(size_t)(mB + r) * F_IN + n] = c[r] + bias;
}

// ---------------------------------------------------------------------------

extern "C" void kernel_launch(void* const* d_in, const int* in_sizes, int n_in,
                              void* d_out, int out_size, void* d_ws, size_t ws_size,
                              hipStream_t stream) {
  const float*     x   = (const float*)d_in[0];
  const long long* ei  = (const long long*)d_in[1];   // [2, E] int64
  const float*     W1  = (const float*)d_in[2];
  const float*     b1  = (const float*)d_in[3];
  const float*     Wf1 = (const float*)d_in[4];
  const float*     bf1 = (const float*)d_in[5];
  const float*     Wf2 = (const float*)d_in[6];
  const float*     bf2 = (const float*)d_in[7];
  float*           out = (float*)d_out;

  // workspace carve-out (256B aligned blocks)
  char* w = (char*)d_ws;
  auto carve = [&](size_t bytes) {
    char* p = w;
    w += (bytes + 255) & ~(size_t)255;
    return p;
  };
  float*  deg   = (float*)carve((size_t)N_NODES * 4);
  float*  dinv  = (float*)carve((size_t)N_NODES * 4);
  __bf16* xb    = (__bf16*)carve((size_t)N_NODES * F_IN * 2);   // 25.6 MB
  __bf16* W1b   = (__bf16*)carve((size_t)F_IN * D1K * 2);
  __bf16* Wf1b  = (__bf16*)carve((size_t)D1K * D2P * 2);
  __bf16* Wf2b  = (__bf16*)carve((size_t)D2P * F_IN * 2);
  __bf16* h     = (__bf16*)carve((size_t)N_NODES * D1P * 2);    // 60.8 MB
  float*  agg   = (float*)carve((size_t)N_NODES * D1P * 4);     // 121.6 MB
  __bf16* aggb  = (__bf16*)carve((size_t)N_NODES * D1K * 2);    // 64 MB
  __bf16* h2b   = xb;   // xb is dead after GEMM1 -> reuse for h2

  // degrees + normalization
  k_init_deg <<<(N_NODES + 255) / 256, 256, 0, stream>>>(deg);
  k_deg_accum<<<(N_EDGES + 255) / 256, 256, 0, stream>>>(ei, deg);
  k_dinv     <<<(N_NODES + 255) / 256, 256, 0, stream>>>(deg, dinv);

  // bf16 conversions
  k_cvt_x  <<<(N_NODES * F_IN + 255) / 256, 256, 0, stream>>>(x, xb);
  k_cvt_W1 <<<(F_IN * D1K + 255) / 256, 256, 0, stream>>>(W1, W1b);
  k_cvt_Wf1<<<(D1K * D2P + 255) / 256, 256, 0, stream>>>(Wf1, Wf1b);
  k_cvt_Wf2<<<(D2P * F_IN + 255) / 256, 256, 0, stream>>>(Wf2, Wf2b);

  // zero accumulator
  {
    size_t n = (size_t)N_NODES * D1P;
    k_zero_agg<<<(unsigned)((n + 255) / 256), 256, 0, stream>>>(agg);
  }

  dim3 wblk(32, 4);                                  // 4 waves, 4 M-tiles
  unsigned gx = (MTILES + 3) / 4;                    // 1563

  // h = x @ W1 (WMMA bf16)
  k_gemm1<<<dim3(gx, (D1P / 16)), wblk, 0, stream>>>(xb, W1b, h);

  // L2-resident edge scatter-add
  k_scatter<<<(N_EDGES + 7) / 8, 256, 0, stream>>>(ei, h, dinv, agg);

  // self-loop + bias + relu -> bf16 K-padded activations
  k_epi1<<<N_NODES, D1K, 0, stream>>>(agg, h, dinv, b1, aggb);

  // MLP head (WMMA bf16)
  k_gemm2<<<dim3(gx, (D2P / 16)), wblk, 0, stream>>>(aggb, Wf1b, bf1, h2b);
  k_gemm3<<<dim3(gx, (F_IN / 16)), wblk, 0, stream>>>(h2b, Wf2b, bf2, out);
}
